// GroupedQueryAttention_15410342658079
// MI455X (gfx1250) — matrix operand order repaired and checked by reading, not compiled, not run on hardware
//
#include <hip/hip_runtime.h>
#include <hip/hip_bf16.h>
#include <math.h>

// ---------------------------------------------------------------------------
// GQA for MI455X (gfx1250, wave32). All matrix math through
// v_wmma_f32_16x16x32_bf16 (bf16 in / f32 accumulate).
// ---------------------------------------------------------------------------

typedef __bf16 bf16;
typedef __attribute__((ext_vector_type(16))) __bf16 v16bf;
typedef __attribute__((ext_vector_type(8)))  __bf16 v8bf;
typedef __attribute__((ext_vector_type(8)))  float  v8f;

#define B_   2
#define T_   2048
#define D_   2048
#define H_   16
#define G_   4
#define HD_  128
#define GS_  (H_ / G_)
#define BT_  (B_ * T_)       // 4096 rows (GEMM M)

static __device__ __forceinline__ v8f wmma_bf16(v16bf a, v16bf b, v8f c) {
    return __builtin_amdgcn_wmma_f32_16x16x32_bf16(
        /*neg_a=*/false, a, /*neg_b=*/false, b,
        /*c_mod=*/(short)0, c, /*reuse_a=*/false, /*reuse_b=*/false);
}

static __device__ __forceinline__ v16bf cat8(v8bf lo, v8bf hi) {
    return __builtin_shufflevector(lo, hi, 0,1,2,3,4,5,6,7,8,9,10,11,12,13,14,15);
}

// ---------------------------------------------------------------------------
// fp32 -> bf16 elementwise conversion (4 elems/thread, memory-bound)
// ---------------------------------------------------------------------------
__global__ void cvt_f32_bf16(const float* __restrict__ src,
                             bf16* __restrict__ dst, int n) {
    int i = (blockIdx.x * blockDim.x + threadIdx.x) * 4;
    if (i + 3 < n) {
        float4 v = *(const float4*)(src + i);
        dst[i + 0] = (bf16)v.x;
        dst[i + 1] = (bf16)v.y;
        dst[i + 2] = (bf16)v.z;
        dst[i + 3] = (bf16)v.w;
    }
}

// ---------------------------------------------------------------------------
// C[M,N] (f32) = A[M,K] (bf16, row-major) @ W[N,K]^T (bf16, row-major)
// Block = 8 waves, wave computes 16(M) x 64(N). K-step = 32 (one WMMA chunk).
// Software-pipelined AND unrolled by 2 with ping-pong phase buffers: each
// iteration computes phase-0 WMMAs while phase-1 fragments load, then
// phase-1 WMMAs while the next phase-0 fragments load. Loads always target
// the buffer not being consumed, so no register-rotation moves are needed
// (the round-2 assembly showed ~30 v_dual_mov per iteration from the
// single-buffer rotation) and WMMAs run under partial loadcnt waits.
// Fragment layouts (contiguous-k order, not the ISA table order; A and B agree):
//   A: lane m = lane&15; half-wave selects contiguous K run {0-15} / {16-31}
//   B: lane n = lane&15; half-wave selects contiguous K run {0-15} / {16-31}
// ---------------------------------------------------------------------------
__global__ __launch_bounds__(256) void gemm_bf16_wmma(
    const bf16* __restrict__ A, const bf16* __restrict__ W,
    float* __restrict__ C, int M, int N, int K) {
    const int lane = threadIdx.x & 31;
    const int wave = threadIdx.x >> 5;
    const int half = lane >> 4;      // 0 / 1
    const int l16  = lane & 15;
    const int m0 = blockIdx.x * 128 + wave * 16;
    const int n0 = blockIdx.y * 64;

    v8f acc[4] = {};
    const bf16* __restrict__ arow  = A + (size_t)(m0 + l16) * K + half * 16;
    const bf16* __restrict__ wrow0 = W + (size_t)(n0 + l16) * K + half * 16;

    auto loadA = [&](int kk) -> v16bf {
        return cat8(*(const v8bf*)(arow + kk), *(const v8bf*)(arow + kk + 8));
    };
    auto loadB = [&](int t, int kk) -> v16bf {
        return *(const v16bf*)(wrow0 + (size_t)t * 16 * K + kk);
    };

    // prologue: phase-0 fragments (kk = 0)
    v16bf a0 = loadA(0);
    v16bf b0[4];
#pragma unroll
    for (int t = 0; t < 4; ++t) b0[t] = loadB(t, 0);

    int kk = 0;
    for (; kk < K - 64; kk += 64) {
        // phase-1 fragments (kk+32) go in flight over phase-0 WMMAs
        v16bf a1 = loadA(kk + 32);
        v16bf b1[4];
#pragma unroll
        for (int t = 0; t < 4; ++t) b1[t] = loadB(t, kk + 32);
        __builtin_prefetch(arow + kk + 512, 0, 3);
#pragma unroll
        for (int t = 0; t < 4; ++t) acc[t] = wmma_bf16(a0, b0[t], acc[t]);
        // next phase-0 fragments (kk+64) go in flight over phase-1 WMMAs
        a0 = loadA(kk + 64);
#pragma unroll
        for (int t = 0; t < 4; ++t) b0[t] = loadB(t, kk + 64);
#pragma unroll
        for (int t = 0; t < 4; ++t) acc[t] = wmma_bf16(a1, b1[t], acc[t]);
    }
    // tail pair: kk == K-64 (phase-0 already resident), load K-32, compute both
    {
        v16bf a1 = loadA(kk + 32);
        v16bf b1[4];
#pragma unroll
        for (int t = 0; t < 4; ++t) b1[t] = loadB(t, kk + 32);
#pragma unroll
        for (int t = 0; t < 4; ++t) acc[t] = wmma_bf16(a0, b0[t], acc[t]);
#pragma unroll
        for (int t = 0; t < 4; ++t) acc[t] = wmma_bf16(a1, b1[t], acc[t]);
    }

#pragma unroll
    for (int t = 0; t < 4; ++t) {
        int col = n0 + t * 16 + l16;
#pragma unroll
        for (int i = 0; i < 8; ++i) {
            int row = m0 + i + half * 8;     // C layout: VGPR i -> M = i (+8 hi half)
            C[(size_t)row * N + col] = acc[t][i];
        }
    }
}

// ---------------------------------------------------------------------------
// Fused RMSNorm + RoPE + scale, fp32 in -> bf16 out, with layout transpose
// in:  [B*T, NH*HD]   out: [B, NH, T, HD]
// One wave per (b, t, head). Lane owns d, d+32, d+64, d+96 so the RoPE
// rotate pair (d <-> d+64) is lane-local.
// ---------------------------------------------------------------------------
__global__ __launch_bounds__(256) void rmsnorm_rope_pack(
    const float* __restrict__ in, bf16* __restrict__ out,
    const float* __restrict__ w, int NH, float scale) {
    const int wid  = blockIdx.x * 8 + (threadIdx.x >> 5);
    const int lane = threadIdx.x & 31;
    const int bt = wid / NH, head = wid % NH;
    const int b = bt / T_, t = bt % T_;

    const float* __restrict__ p = in + (size_t)bt * NH * HD_ + head * HD_;
    float e0 = p[lane], e1 = p[lane + 32], e2 = p[lane + 64], e3 = p[lane + 96];
    float ss = e0 * e0 + e1 * e1 + e2 * e2 + e3 * e3;
#pragma unroll
    for (int m = 16; m >= 1; m >>= 1) ss += __shfl_xor(ss, m, 32);
    float r = rsqrtf(ss * (1.0f / HD_) + 1e-6f);
    float n0 = e0 * r * w[lane],      n1 = e1 * r * w[lane + 32];
    float n2 = e2 * r * w[lane + 64], n3 = e3 * r * w[lane + 96];

    // RoPE: inv_freq[p] = 10000^(-2p/128); cos/sin repeat with period 64
    const float LOG_BASE = 9.2103403719761836f;  // ln(10000)
    float tt = (float)t;
    float f0 = tt * __expf(-((float)(2 * lane)        / HD_) * LOG_BASE);
    float f1 = tt * __expf(-((float)(2 * (lane + 32)) / HD_) * LOG_BASE);
    float s0, c0, s1, c1;
    __sincosf(f0, &s0, &c0);
    __sincosf(f1, &s1, &c1);
    float o0 = n0 * c0 - n2 * s0;
    float o1 = n1 * c1 - n3 * s1;
    float o2 = n2 * c0 + n0 * s0;
    float o3 = n3 * c1 + n1 * s1;

    bf16* __restrict__ q = out + (((size_t)b * NH + head) * T_ + t) * HD_;
    q[lane]      = (bf16)(o0 * scale);
    q[lane + 32] = (bf16)(o1 * scale);
    q[lane + 64] = (bf16)(o2 * scale);
    q[lane + 96] = (bf16)(o3 * scale);
}

// ---------------------------------------------------------------------------
// V: fp32 [B*T, G*HD] -> bf16 [B, G, HD, T]  (transposed for PV B-fragments)
// ---------------------------------------------------------------------------
__global__ void v_transpose_bf16(const float* __restrict__ Vf,
                                 bf16* __restrict__ Vt, int total) {
    int idx = blockIdx.x * blockDim.x + threadIdx.x;
    if (idx >= total) return;
    int d = idx & (HD_ - 1);
    int g = (idx >> 7) & (G_ - 1);
    int t = (idx >> 9) & (T_ - 1);
    int b = idx >> 20;             // 512 * 2048 = 2^20 elements per batch
    Vt[(((size_t)(b * G_ + g) * HD_) + d) * T_ + t] = (bf16)Vf[idx];
}

// ---------------------------------------------------------------------------
// Flash attention, one wave per (b, h, 16-query tile), causal, online softmax.
// Q: [B,H,T,HD] bf16 (pre-scaled by 1/HD), K: [B,G,T,HD], V: [B,G,HD,T].
// Per 32-key step: 8 WMMAs (scores, K=128) + 8 WMMAs (P@V, N=128).
// P transposed C-layout -> A-layout through a 1 KB LDS tile.
// ---------------------------------------------------------------------------
__global__ __launch_bounds__(32) void flash_attn_wmma(
    const bf16* __restrict__ Q, const bf16* __restrict__ Kc,
    const bf16* __restrict__ Vt, bf16* __restrict__ ctx) {
    __shared__ bf16 sp[16 * 32];
    const int lane = threadIdx.x & 31;
    const int half = lane >> 4;
    const int l16  = lane & 15;
    const int q0 = blockIdx.x * 16;
    const int h  = blockIdx.y;
    const int b  = blockIdx.z;
    const int g  = h / GS_;

    // Q A-fragments: 4 chunks of K=32 covering HD=128
    const bf16* __restrict__ qrow =
        Q + (size_t)((b * H_ + h) * T_ + q0 + l16) * HD_;
    v16bf qf[4];
#pragma unroll
    for (int c = 0; c < 4; ++c) {
        v8bf lo = *(const v8bf*)(qrow + c * 32 + half * 16);
        v8bf hi = *(const v8bf*)(qrow + c * 32 + 8 + half * 16);
        qf[c] = cat8(lo, hi);
    }

    v8f o[8] = {};
    float mrow[8], lrow[8];
#pragma unroll
    for (int i = 0; i < 8; ++i) { mrow[i] = -3.0e38f; lrow[i] = 0.0f; }

    const bf16* __restrict__ kbase = Kc + (size_t)(b * G_ + g) * T_ * HD_;
    const bf16* __restrict__ vbase = Vt + (size_t)(b * G_ + g) * HD_ * T_;

    for (int j0 = 0; j0 < q0 + 16; j0 += 32) {
        // ---- scores S[16q x 32k] = Q @ K^T (K-dim = HD = 128) ----
        v8f s0 = {}, s1 = {};
        const bf16* k0 = kbase + (size_t)(j0 + l16) * HD_ + half * 16;
        const bf16* k1 = kbase + (size_t)(j0 + 16 + l16) * HD_ + half * 16;
#pragma unroll
        for (int c = 0; c < 4; ++c) {
            s0 = wmma_bf16(qf[c], *(const v16bf*)(k0 + c * 32), s0);
            s1 = wmma_bf16(qf[c], *(const v16bf*)(k1 + c * 32), s1);
        }
        // ---- causal mask + online softmax (C layout: row = i + 8*half) ----
#pragma unroll
        for (int i = 0; i < 8; ++i) {
            int row = q0 + i + half * 8;
            float a = (j0 + l16      <= row) ? s0[i] : -3.0e38f;
            float c = (j0 + 16 + l16 <= row) ? s1[i] : -3.0e38f;
            float t = fmaxf(a, c);
            t = fmaxf(t, __shfl_xor(t, 1, 32));
            t = fmaxf(t, __shfl_xor(t, 2, 32));
            t = fmaxf(t, __shfl_xor(t, 4, 32));
            t = fmaxf(t, __shfl_xor(t, 8, 32));
            float mnew  = fmaxf(mrow[i], t);
            float alpha = __expf(mrow[i] - mnew);
            float p0 = __expf(a - mnew);
            float p1 = __expf(c - mnew);
            float rs = p0 + p1;
            rs += __shfl_xor(rs, 1, 32);
            rs += __shfl_xor(rs, 2, 32);
            rs += __shfl_xor(rs, 4, 32);
            rs += __shfl_xor(rs, 8, 32);
            lrow[i] = lrow[i] * alpha + rs;
            mrow[i] = mnew;
#pragma unroll
            for (int dt = 0; dt < 8; ++dt) o[dt][i] *= alpha;
            sp[(i + half * 8) * 32 + l16]      = (bf16)p0;
            sp[(i + half * 8) * 32 + 16 + l16] = (bf16)p1;
        }
        asm volatile("s_wait_dscnt 0" ::: "memory");  // LDS transpose hazard
        __syncthreads();                               // single wave -> S_NOP
        // ---- reload P in A-fragment layout ----
        v8bf plo = *(const v8bf*)(&sp[l16 * 32 + half * 16]);
        v8bf phi = *(const v8bf*)(&sp[l16 * 32 + 8 + half * 16]);
        v16bf pa = cat8(plo, phi);
        // ---- O += P @ V  (B-fragment: contiguous 16 keys per lane in Vt) ----
        const bf16* vrow = vbase + j0 + half * 16;
#pragma unroll
        for (int dt = 0; dt < 8; ++dt) {
            v16bf vb = *(const v16bf*)(vrow + (size_t)(dt * 16 + l16) * T_);
            o[dt] = wmma_bf16(pa, vb, o[dt]);
        }
    }

    // epilogue: normalize and store ctx bf16 [B*T, H*HD]
#pragma unroll
    for (int dt = 0; dt < 8; ++dt) {
        int col = h * HD_ + dt * 16 + l16;
#pragma unroll
        for (int i = 0; i < 8; ++i) {
            int row = q0 + i + half * 8;
            ctx[(size_t)(b * T_ + row) * (H_ * HD_) + col] =
                (bf16)(o[dt][i] * (1.0f / lrow[i]));
        }
    }
}

// ---------------------------------------------------------------------------
// Host-side orchestration
// ---------------------------------------------------------------------------
static inline size_t align256(size_t x) { return (x + 255) & ~(size_t)255; }

extern "C" void kernel_launch(void* const* d_in, const int* in_sizes, int n_in,
                              void* d_out, int out_size, void* d_ws, size_t ws_size,
                              hipStream_t stream) {
    const float* x  = (const float*)d_in[0];
    const float* Wq = (const float*)d_in[1];
    const float* Wk = (const float*)d_in[2];
    const float* Wv = (const float*)d_in[3];
    const float* Wo = (const float*)d_in[4];
    const float* qn = (const float*)d_in[5];
    const float* kn = (const float*)d_in[6];
    float* out = (float*)d_out;

    // workspace carve
    char* w = (char*)d_ws;
    size_t off = 0;
    auto take = [&](size_t bytes) { void* p = w + off; off += align256(bytes); return p; };
    const size_t nX  = (size_t)BT_ * D_;        // 8388608
    const size_t nWq = (size_t)D_ * D_;         // 4194304
    const size_t nWk = (size_t)(G_ * HD_) * D_; // 1048576
    bf16*  xb   = (bf16*)take(nX * 2);
    bf16*  wqb  = (bf16*)take(nWq * 2);
    bf16*  wkb  = (bf16*)take(nWk * 2);
    bf16*  wvb  = (bf16*)take(nWk * 2);
    bf16*  wob  = (bf16*)take(nWq * 2);
    float* Qf   = (float*)take((size_t)BT_ * D_ * 4);
    float* Kf   = (float*)take((size_t)BT_ * (G_ * HD_) * 4);
    float* Vf   = (float*)take((size_t)BT_ * (G_ * HD_) * 4);
    bf16*  qb   = (bf16*)take((size_t)B_ * H_ * T_ * HD_ * 2);
    bf16*  kb   = (bf16*)take((size_t)B_ * G_ * T_ * HD_ * 2);
    bf16*  vt   = (bf16*)take((size_t)B_ * G_ * T_ * HD_ * 2);
    bf16*  ctxb = (bf16*)take((size_t)BT_ * D_ * 2);
    (void)ws_size; (void)in_sizes; (void)n_in; (void)out_size;

    // 1) fp32 -> bf16 conversions
    cvt_f32_bf16<<<(int)(nX  / 1024), 256, 0, stream>>>(x,  xb,  (int)nX);
    cvt_f32_bf16<<<(int)(nWq / 1024), 256, 0, stream>>>(Wq, wqb, (int)nWq);
    cvt_f32_bf16<<<(int)(nWk / 1024), 256, 0, stream>>>(Wk, wkb, (int)nWk);
    cvt_f32_bf16<<<(int)(nWk / 1024), 256, 0, stream>>>(Wv, wvb, (int)nWk);
    cvt_f32_bf16<<<(int)(nWq / 1024), 256, 0, stream>>>(Wo, wob, (int)nWq);

    // 2) projections (WMMA GEMMs): [4096,2048] @ W^T
    gemm_bf16_wmma<<<dim3(BT_ / 128, D_ / 64), 256, 0, stream>>>(
        xb, wqb, Qf, BT_, D_, D_);
    gemm_bf16_wmma<<<dim3(BT_ / 128, (G_ * HD_) / 64), 256, 0, stream>>>(
        xb, wkb, Kf, BT_, G_ * HD_, D_);
    gemm_bf16_wmma<<<dim3(BT_ / 128, (G_ * HD_) / 64), 256, 0, stream>>>(
        xb, wvb, Vf, BT_, G_ * HD_, D_);

    // 3) RMSNorm + RoPE (+ fold score scale 1/HD into q)
    rmsnorm_rope_pack<<<BT_ * H_ / 8, 256, 0, stream>>>(Qf, qb, qn, H_, 1.0f / HD_);
    rmsnorm_rope_pack<<<BT_ * G_ / 8, 256, 0, stream>>>(Kf, kb, kn, G_, 1.0f);

    // 4) V transpose to [B,G,HD,T]
    v_transpose_bf16<<<(int)((size_t)BT_ * G_ * HD_ / 256), 256, 0, stream>>>(
        Vf, vt, BT_ * G_ * HD_);

    // 5) flash attention (WMMA scores + WMMA P@V)
    flash_attn_wmma<<<dim3(T_ / 16, H_, B_), 32, 0, stream>>>(qb, kb, vt, ctxb);

    // 6) output projection straight into d_out (f32)
    gemm_bf16_wmma<<<dim3(BT_ / 128, D_ / 64), 256, 0, stream>>>(
        ctxb, wob, out, BT_, D_, D_);
}
